// MAM_47691316855321
// MI455X (gfx1250) — matrix operand order repaired and checked by reading, not compiled, not run on hardware
//
#include <hip/hip_runtime.h>

#define DIM     1024
#define LSEQ    5
#define NROWS   81920          // 16384 * 5
#define BM      64             // rows per workgroup (4 M-tiles)
#define NMT     4              // M-tiles per block
#define NTW     4              // N-tiles per wave (64 cols); 16 waves -> 1024
#define KSTEP   32
#define NKT     (DIM / KSTEP)  // 32 k-steps
#define THREADS 512            // 16 waves
#define SMEM_BYTES (BM * DIM * 2)   // 128 KB dynamic LDS

typedef __attribute__((ext_vector_type(16))) __bf16 bf16x16;
typedef __attribute__((ext_vector_type(8)))  float  f32x8;
typedef unsigned int u32x4 __attribute__((ext_vector_type(4)));
typedef int          i32x8 __attribute__((ext_vector_type(8)));
typedef int          i32x4 __attribute__((ext_vector_type(4)));

#if __has_builtin(__builtin_amdgcn_tensor_load_to_lds) && \
    __has_builtin(__builtin_amdgcn_s_wait_tensorcnt)
#define HAVE_TDM 1
#else
#define HAVE_TDM 0
#endif

union FragU {
    unsigned int u[8];
    uint4        q[2];
    bf16x16      v;
};

__device__ __forceinline__ unsigned short f2bf(float f) {
    unsigned int u = __float_as_uint(f);
    u += 0x7FFFu + ((u >> 16) & 1u);           // round-to-nearest-even
    return (unsigned short)(u >> 16);
}
__device__ __forceinline__ float bf2f(unsigned short h) {
    return __uint_as_float(((unsigned int)h) << 16);
}

__device__ __forceinline__ f32x8 zero8() {
    f32x8 z = {0.f, 0.f, 0.f, 0.f, 0.f, 0.f, 0.f, 0.f};
    return z;
}

__device__ __forceinline__ f32x8 wmma_bf16(bf16x16 a, bf16x16 b, f32x8 c) {
    return __builtin_amdgcn_wmma_f32_16x16x32_bf16(false, a, false, b,
                                                   (short)0, c, false, false);
}

// A fragment (16x32 bf16) from LDS row-major tile.
__device__ __forceinline__ bf16x16 load_a(const unsigned short* tile, int mt,
                                          int kb, int lane) {
    FragU f;
    const unsigned short* p =
        tile + (mt * 16 + (lane & 15)) * DIM + kb + ((lane & 16) ? 8 : 0);
    f.q[0] = *(const uint4*)(p);        // K = +0..7
    f.q[1] = *(const uint4*)(p + 16);   // K = +16..23
    return f.v;
}

// B fragment (32x16 bf16) == W^T tile; W row-major [n][k].
__device__ __forceinline__ bf16x16 load_b(const unsigned short* __restrict__ W,
                                          int nb, int kb, int lane) {
    FragU f;
    const unsigned short* p =
        W + (size_t)(nb + (lane & 15)) * DIM + kb + ((lane & 16) ? 8 : 0);
    f.q[0] = *(const uint4*)(p);
    f.q[1] = *(const uint4*)(p + 16);
    return f.v;
}

__device__ __forceinline__ void stage_f32(const float* __restrict__ src,
                                          unsigned short* tile, int tid) {
    const float4* s4 = (const float4*)src;
#pragma unroll 4
    for (int i = 0; i < (BM * DIM / 4) / THREADS; ++i) {
        int idx = tid + i * THREADS;
        float4 f = s4[idx];
        union { unsigned short us[4]; uint2 u2; } p;
        p.us[0] = f2bf(f.x); p.us[1] = f2bf(f.y);
        p.us[2] = f2bf(f.z); p.us[3] = f2bf(f.w);
        *(uint2*)(tile + 4 * idx) = p.u2;
    }
}

__device__ __forceinline__ void stage_bf(const unsigned short* __restrict__ src,
                                         unsigned short* tile, int tid) {
#if HAVE_TDM
    // Tensor Data Mover: one DMA moves the whole contiguous 128KB tile
    // (16384 x 8-byte elements, 1-D) from global to LDS. Wave 0 issues it
    // and waits on TENSORcnt; everyone else just waits at the barrier.
    if (tid < 32) {
        unsigned long long ga = (unsigned long long)(uintptr_t)src;
        unsigned int lds = (unsigned int)(uintptr_t)tile;
        u32x4 g0 = {1u,                                  // count=1, user D#
                    lds,                                 // lds_addr
                    (unsigned int)(ga & 0xFFFFFFFFull),  // global_addr lo
                    (unsigned int)(((ga >> 32) & 0x01FFFFFFull) |
                                   0x80000000u)};        // ga hi | type=2
        i32x8 g1 = {(int)0x00030000,  // wg_mask=0, data_size=3 (8B)
                    (int)0x40000000,  // tensor_dim0 = 16384 (bits 79:48)
                    (int)0x00010000,  // tensor_dim1 = 1     (bits 111:80)
                    (int)0x40000000,  // tile_dim0   = 16384 (bits 127:112)
                    0,                // tile_dim1/2 unused
                    16384,            // tensor_dim0_stride lo
                    0, 0};
        i32x4 g2 = {0, 0, 0, 0};
        i32x4 g3 = {0, 0, 0, 0};
        i32x8 g4 = {0, 0, 0, 0, 0, 0, 0, 0};
        __builtin_amdgcn_tensor_load_to_lds(g0, g1, g2, g3, g4, 0);
        __builtin_amdgcn_s_wait_tensorcnt(0);
    }
#else
    const uint4* s = (const uint4*)src;
    uint4*       d = (uint4*)tile;
#pragma unroll 4
    for (int i = 0; i < (BM * DIM) / (THREADS * 8); ++i)
        d[tid + i * THREADS] = s[tid + i * THREADS];
#endif
}

// ---------------------------------------------------------------- weights cvt
__global__ __launch_bounds__(THREADS) void k_cvt(const float* __restrict__ src,
                                                 unsigned short* __restrict__ dst,
                                                 int n) {
    int i = blockIdx.x * THREADS + threadIdx.x;
    if (i < n) dst[i] = f2bf(src[i]);
}

// ------------------------------------------------------- scores = q.k / 32
__global__ __launch_bounds__(THREADS) void k_scores(
    const float* __restrict__ x, const unsigned short* __restrict__ Wq,
    const unsigned short* __restrict__ Wk, float* __restrict__ scores) {
    extern __shared__ __align__(16) unsigned char smem[];
    unsigned short* tile = (unsigned short*)smem;
    const int tid = threadIdx.x, lane = tid & 31, w = tid >> 5;
    const int row0 = blockIdx.x * BM;

    stage_f32(x + (size_t)row0 * DIM, tile, tid);
    __syncthreads();

    float part[NMT][8];
#pragma unroll
    for (int mt = 0; mt < NMT; ++mt)
#pragma unroll
        for (int vi = 0; vi < 8; ++vi) part[mt][vi] = 0.f;

    for (int j = 0; j < NTW; ++j) {
        const int nb = w * (NTW * 16) + j * 16;
        f32x8 cq[NMT], ck[NMT];
#pragma unroll
        for (int mt = 0; mt < NMT; ++mt) { cq[mt] = zero8(); ck[mt] = zero8(); }
        for (int kt = 0; kt < NKT; ++kt) {
            const int kb = kt * KSTEP;
            __builtin_prefetch(Wq + (size_t)(nb + (lane & 15)) * DIM + kb + 128,
                               0, 3);
            bf16x16 a[NMT];
#pragma unroll
            for (int mt = 0; mt < NMT; ++mt) a[mt] = load_a(tile, mt, kb, lane);
            bf16x16 bq = load_b(Wq, nb, kb, lane);
#pragma unroll
            for (int mt = 0; mt < NMT; ++mt) cq[mt] = wmma_bf16(a[mt], bq, cq[mt]);
            bf16x16 bk = load_b(Wk, nb, kb, lane);
#pragma unroll
            for (int mt = 0; mt < NMT; ++mt) ck[mt] = wmma_bf16(a[mt], bk, ck[mt]);
        }
#pragma unroll
        for (int mt = 0; mt < NMT; ++mt)
#pragma unroll
            for (int vi = 0; vi < 8; ++vi) part[mt][vi] += cq[mt][vi] * ck[mt][vi];
    }

    __syncthreads();                // LDS tile no longer needed; reuse for sums
    float* ssum = (float*)smem;
    if (tid < BM) ssum[tid] = 0.f;
    __syncthreads();
#pragma unroll
    for (int mt = 0; mt < NMT; ++mt)
#pragma unroll
        for (int vi = 0; vi < 8; ++vi) {
            float v = part[mt][vi];
#pragma unroll
            for (int d = 1; d < 16; d <<= 1) v += __shfl_xor(v, d, 32);
            if ((lane & 15) == 0)
                atomicAdd(&ssum[mt * 16 + vi + ((lane & 16) ? 8 : 0)], v);
        }
    __syncthreads();
    if (tid < BM) scores[row0 + tid] = ssum[tid] * 0.03125f;  // 1/sqrt(1024)
}

// ------------------------------- v GEMM + masked softmax + residual + LN -> y
__global__ __launch_bounds__(THREADS) void k_attn_ln(
    const float* __restrict__ x, const unsigned short* __restrict__ Wv,
    const float* __restrict__ scores, const long long* __restrict__ mask,
    const float* __restrict__ gam, const float* __restrict__ bet,
    unsigned short* __restrict__ y_bf) {
    extern __shared__ __align__(16) unsigned char smem[];
    unsigned short* tile = (unsigned short*)smem;
    const int tid = threadIdx.x, lane = tid & 31, w = tid >> 5;
    const int row0 = blockIdx.x * BM;

    stage_f32(x + (size_t)row0 * DIM, tile, tid);
    __syncthreads();

    f32x8 acc[NMT][NTW];
#pragma unroll
    for (int mt = 0; mt < NMT; ++mt)
#pragma unroll
        for (int j = 0; j < NTW; ++j) acc[mt][j] = zero8();

    for (int kt = 0; kt < NKT; ++kt) {
        const int kb = kt * KSTEP;
        __builtin_prefetch(Wv + (size_t)(w * (NTW * 16) + lane) * DIM + kb + 128,
                           0, 3);
        bf16x16 a[NMT];
#pragma unroll
        for (int mt = 0; mt < NMT; ++mt) a[mt] = load_a(tile, mt, kb, lane);
#pragma unroll
        for (int j = 0; j < NTW; ++j) {
            bf16x16 b = load_b(Wv, w * (NTW * 16) + j * 16, kb, lane);
#pragma unroll
            for (int mt = 0; mt < NMT; ++mt)
                acc[mt][j] = wmma_bf16(a[mt], b, acc[mt][j]);
        }
    }

    __syncthreads();                // reuse LDS for probs + LN stats
    float* sprob = (float*)smem;
    float* srsum = sprob + BM;
    float* srsq  = srsum + BM;
    float* smean = srsq + BM;
    float* srstd = smean + BM;
    if (tid < BM) {
        const int row = row0 + tid;
        const int b = row / LSEQ;
        const int l = row - b * LSEQ;
        float m = -3.0e38f;
        bool any = false;
        float sv[LSEQ];
        bool  kp[LSEQ];
#pragma unroll
        for (int ll = 0; ll < LSEQ; ++ll) {
            kp[ll] = (mask[ll] != 0);
            sv[ll] = kp[ll] ? scores[b * LSEQ + ll] : 0.f;
            if (kp[ll]) { m = fmaxf(m, sv[ll]); any = true; }
        }
        float sum = 0.f, ev = 0.f;
#pragma unroll
        for (int ll = 0; ll < LSEQ; ++ll) {
            float e = kp[ll] ? __expf(sv[ll] - m) : 0.f;
            if (ll == l) ev = e;
            sum += e;
        }
        sprob[tid] = (any && sum > 0.f) ? (ev / sum) : 0.f;
        srsum[tid] = 0.f;
        srsq[tid]  = 0.f;
    }
    __syncthreads();

#pragma unroll
    for (int mt = 0; mt < NMT; ++mt)
#pragma unroll
        for (int j = 0; j < NTW; ++j) {
            const int col = w * (NTW * 16) + j * 16 + (lane & 15);
            const int rb0 = mt * 16 + ((lane & 16) ? 8 : 0);
#pragma unroll
            for (int vi = 0; vi < 8; ++vi) {
                const int rib = rb0 + vi;
                const int row = row0 + rib;
                float t = sprob[rib] * acc[mt][j][vi] +
                          x[(size_t)row * DIM + col];
                acc[mt][j][vi] = t;
                float ts = t, tq = t * t;
#pragma unroll
                for (int d = 1; d < 16; d <<= 1) {
                    ts += __shfl_xor(ts, d, 32);
                    tq += __shfl_xor(tq, d, 32);
                }
                if ((lane & 15) == 0) {
                    atomicAdd(&srsum[rib], ts);
                    atomicAdd(&srsq[rib], tq);
                }
            }
        }
    __syncthreads();
    if (tid < BM) {
        float mu  = srsum[tid] * (1.f / DIM);
        float var = srsq[tid] * (1.f / DIM) - mu * mu;
        smean[tid] = mu;
        srstd[tid] = rsqrtf(var + 1e-5f);
    }
    __syncthreads();
#pragma unroll
    for (int mt = 0; mt < NMT; ++mt)
#pragma unroll
        for (int j = 0; j < NTW; ++j) {
            const int col = w * (NTW * 16) + j * 16 + (lane & 15);
            const int rb0 = mt * 16 + ((lane & 16) ? 8 : 0);
#pragma unroll
            for (int vi = 0; vi < 8; ++vi) {
                const int rib = rb0 + vi;
                const int row = row0 + rib;
                float yv = (acc[mt][j][vi] - smean[rib]) * srstd[rib] *
                               gam[col] + bet[col];
                y_bf[(size_t)row * DIM + col] = f2bf(yv);
            }
        }
}

// ------------------------------------------- h = gelu(y @ W1^T + b1) (bf16)
__global__ __launch_bounds__(THREADS) void k_ffn1(
    const unsigned short* __restrict__ y_bf,
    const unsigned short* __restrict__ W1, const float* __restrict__ b1,
    unsigned short* __restrict__ h_bf) {
    extern __shared__ __align__(16) unsigned char smem[];
    unsigned short* tile = (unsigned short*)smem;
    const int tid = threadIdx.x, lane = tid & 31, w = tid >> 5;
    const int row0 = blockIdx.x * BM;

    stage_bf(y_bf + (size_t)row0 * DIM, tile, tid);
    __syncthreads();

    f32x8 acc[NMT][NTW];
#pragma unroll
    for (int mt = 0; mt < NMT; ++mt)
#pragma unroll
        for (int j = 0; j < NTW; ++j) acc[mt][j] = zero8();

    for (int kt = 0; kt < NKT; ++kt) {
        const int kb = kt * KSTEP;
        __builtin_prefetch(W1 + (size_t)(w * (NTW * 16) + lane) * DIM + kb + 128,
                           0, 3);
        bf16x16 a[NMT];
#pragma unroll
        for (int mt = 0; mt < NMT; ++mt) a[mt] = load_a(tile, mt, kb, lane);
#pragma unroll
        for (int j = 0; j < NTW; ++j) {
            bf16x16 b = load_b(W1, w * (NTW * 16) + j * 16, kb, lane);
#pragma unroll
            for (int mt = 0; mt < NMT; ++mt)
                acc[mt][j] = wmma_bf16(a[mt], b, acc[mt][j]);
        }
    }

#pragma unroll
    for (int mt = 0; mt < NMT; ++mt)
#pragma unroll
        for (int j = 0; j < NTW; ++j) {
            const int col = w * (NTW * 16) + j * 16 + (lane & 15);
            const int rb0 = mt * 16 + ((lane & 16) ? 8 : 0);
            const float bb = b1[col];
#pragma unroll
            for (int vi = 0; vi < 8; ++vi) {
                const int row = row0 + rb0 + vi;
                float v = acc[mt][j][vi] + bb;
                float g = 0.5f * v * (1.f + erff(v * 0.70710678118654752f));
                h_bf[(size_t)row * DIM + col] = f2bf(g);
            }
        }
}

// --------------------------- out = LN(h @ W2^T + b2 + y) * ff_g + ff_b (f32)
__global__ __launch_bounds__(THREADS) void k_ffn2(
    const unsigned short* __restrict__ h_bf,
    const unsigned short* __restrict__ W2, const float* __restrict__ b2,
    const unsigned short* __restrict__ y_bf, const float* __restrict__ gam,
    const float* __restrict__ bet, float* __restrict__ out) {
    extern __shared__ __align__(16) unsigned char smem[];
    unsigned short* tile = (unsigned short*)smem;
    const int tid = threadIdx.x, lane = tid & 31, w = tid >> 5;
    const int row0 = blockIdx.x * BM;

    stage_bf(h_bf + (size_t)row0 * DIM, tile, tid);
    __syncthreads();

    f32x8 acc[NMT][NTW];
#pragma unroll
    for (int mt = 0; mt < NMT; ++mt)
#pragma unroll
        for (int j = 0; j < NTW; ++j) acc[mt][j] = zero8();

    for (int kt = 0; kt < NKT; ++kt) {
        const int kb = kt * KSTEP;
        __builtin_prefetch(W2 + (size_t)(w * (NTW * 16) + lane) * DIM + kb + 128,
                           0, 3);
        bf16x16 a[NMT];
#pragma unroll
        for (int mt = 0; mt < NMT; ++mt) a[mt] = load_a(tile, mt, kb, lane);
#pragma unroll
        for (int j = 0; j < NTW; ++j) {
            bf16x16 b = load_b(W2, w * (NTW * 16) + j * 16, kb, lane);
#pragma unroll
            for (int mt = 0; mt < NMT; ++mt)
                acc[mt][j] = wmma_bf16(a[mt], b, acc[mt][j]);
        }
    }

    __syncthreads();                // reuse LDS for LN stats
    float* srsum = (float*)smem;
    float* srsq  = srsum + BM;
    float* smean = srsq + BM;
    float* srstd = smean + BM;
    if (tid < BM) { srsum[tid] = 0.f; srsq[tid] = 0.f; }
    __syncthreads();

#pragma unroll
    for (int mt = 0; mt < NMT; ++mt)
#pragma unroll
        for (int j = 0; j < NTW; ++j) {
            const int col = w * (NTW * 16) + j * 16 + (lane & 15);
            const int rb0 = mt * 16 + ((lane & 16) ? 8 : 0);
            const float bb = b2[col];
#pragma unroll
            for (int vi = 0; vi < 8; ++vi) {
                const int rib = rb0 + vi;
                const int row = row0 + rib;
                float t = acc[mt][j][vi] + bb +
                          bf2f(y_bf[(size_t)row * DIM + col]);
                acc[mt][j][vi] = t;
                float ts = t, tq = t * t;
#pragma unroll
                for (int d = 1; d < 16; d <<= 1) {
                    ts += __shfl_xor(ts, d, 32);
                    tq += __shfl_xor(tq, d, 32);
                }
                if ((lane & 15) == 0) {
                    atomicAdd(&srsum[rib], ts);
                    atomicAdd(&srsq[rib], tq);
                }
            }
        }
    __syncthreads();
    if (tid < BM) {
        float mu  = srsum[tid] * (1.f / DIM);
        float var = srsq[tid] * (1.f / DIM) - mu * mu;
        smean[tid] = mu;
        srstd[tid] = rsqrtf(var + 1e-5f);
    }
    __syncthreads();
#pragma unroll
    for (int mt = 0; mt < NMT; ++mt)
#pragma unroll
        for (int j = 0; j < NTW; ++j) {
            const int col = w * (NTW * 16) + j * 16 + (lane & 15);
            const int rb0 = mt * 16 + ((lane & 16) ? 8 : 0);
#pragma unroll
            for (int vi = 0; vi < 8; ++vi) {
                const int rib = rb0 + vi;
                const int row = row0 + rib;
                out[(size_t)row * DIM + col] =
                    (acc[mt][j][vi] - smean[rib]) * srstd[rib] * gam[col] +
                    bet[col];
            }
        }
}

extern "C" void kernel_launch(void* const* d_in, const int* in_sizes, int n_in,
                              void* d_out, int out_size, void* d_ws,
                              size_t ws_size, hipStream_t stream) {
    const float*     x     = (const float*)d_in[0];
    const long long* mask  = (const long long*)d_in[1];
    const float*     Wq    = (const float*)d_in[2];
    const float*     Wk    = (const float*)d_in[3];
    const float*     Wv    = (const float*)d_in[4];
    const float*     W1    = (const float*)d_in[5];
    const float*     b1    = (const float*)d_in[6];
    const float*     W2    = (const float*)d_in[7];
    const float*     b2    = (const float*)d_in[8];
    const float*     att_g = (const float*)d_in[9];
    const float*     att_b = (const float*)d_in[10];
    const float*     ff_g  = (const float*)d_in[11];
    const float*     ff_b  = (const float*)d_in[12];
    float*           out   = (float*)d_out;

    const size_t MB = 1024ull * 1024ull;
    unsigned char* ws = (unsigned char*)d_ws;
    unsigned short* wq_bf = (unsigned short*)(ws + 0 * MB);
    unsigned short* wk_bf = (unsigned short*)(ws + 2 * MB);
    unsigned short* wv_bf = (unsigned short*)(ws + 4 * MB);
    unsigned short* w1_bf = (unsigned short*)(ws + 6 * MB);
    unsigned short* w2_bf = (unsigned short*)(ws + 8 * MB);
    float*          sc    = (float*)(ws + 10 * MB);
    unsigned short* y_bf  = (unsigned short*)(ws + 11 * MB);
    unsigned short* h_bf  = (unsigned short*)(ws + 11 * MB +
                                              (size_t)NROWS * DIM * 2);

    const int nW = DIM * DIM;
    const int cvtBlocks = (nW + THREADS - 1) / THREADS;
    k_cvt<<<cvtBlocks, THREADS, 0, stream>>>(Wq, wq_bf, nW);
    k_cvt<<<cvtBlocks, THREADS, 0, stream>>>(Wk, wk_bf, nW);
    k_cvt<<<cvtBlocks, THREADS, 0, stream>>>(Wv, wv_bf, nW);
    k_cvt<<<cvtBlocks, THREADS, 0, stream>>>(W1, w1_bf, nW);
    k_cvt<<<cvtBlocks, THREADS, 0, stream>>>(W2, w2_bf, nW);

    const int gemmBlocks = NROWS / BM;  // 1280
    k_scores<<<gemmBlocks, THREADS, SMEM_BYTES, stream>>>(x, wq_bf, wk_bf, sc);
    k_attn_ln<<<gemmBlocks, THREADS, SMEM_BYTES, stream>>>(x, wv_bf, sc, mask,
                                                           att_g, att_b, y_bf);
    k_ffn1<<<gemmBlocks, THREADS, SMEM_BYTES, stream>>>(y_bf, w1_bf, b1, h_bf);
    k_ffn2<<<gemmBlocks, THREADS, SMEM_BYTES, stream>>>(h_bf, w2_bf, b2, y_bf,
                                                        ff_g, ff_b, out);
}
